// MultiHeadAttention_16192026706019
// MI455X (gfx1250) — matrix operand order repaired and checked by reading, not compiled, not run on hardware
//
#include <hip/hip_runtime.h>
#include <hip/hip_bf16.h>
#include <math.h>

typedef __attribute__((ext_vector_type(16))) __bf16 v16bf;
typedef __attribute__((ext_vector_type(8)))  float  v8f;

#define BB  4
#define SS  2048
#define DD  768
#define HH  12
#define DHH 64

union BF16x16 { v16bf v; uint4 q[2]; };

__device__ inline v8f wmma_bf16(v16bf a, v16bf b, v8f c) {
  return __builtin_amdgcn_wmma_f32_16x16x32_bf16(false, a, false, b, (short)0, c, false, false);
}

// A-fragment from f32: elements 0..7 = K base..base+7, elements 8..15 = K base+16..base+23
__device__ inline v16bf load_cvt_A(const float* p) {
  float4 f0 = *reinterpret_cast<const float4*>(p);
  float4 f1 = *reinterpret_cast<const float4*>(p + 4);
  float4 f2 = *reinterpret_cast<const float4*>(p + 16);
  float4 f3 = *reinterpret_cast<const float4*>(p + 20);
  v16bf v;
  v[0]=(__bf16)f0.x;  v[1]=(__bf16)f0.y;  v[2]=(__bf16)f0.z;  v[3]=(__bf16)f0.w;
  v[4]=(__bf16)f1.x;  v[5]=(__bf16)f1.y;  v[6]=(__bf16)f1.z;  v[7]=(__bf16)f1.w;
  v[8]=(__bf16)f2.x;  v[9]=(__bf16)f2.y;  v[10]=(__bf16)f2.z; v[11]=(__bf16)f2.w;
  v[12]=(__bf16)f3.x; v[13]=(__bf16)f3.y; v[14]=(__bf16)f3.z; v[15]=(__bf16)f3.w;
  return v;
}

// f32 -> bf16 bulk conversion (4 elements/thread)
__global__ __launch_bounds__(256) void cvt4(const float4* __restrict__ in,
                                            uint2* __restrict__ out, int n4) {
  int i = blockIdx.x * blockDim.x + threadIdx.x;
  if (i < n4) {
    float4 f = in[i];
    union { __bf16 h[4]; uint2 u; } u;
    u.h[0] = (__bf16)f.x; u.h[1] = (__bf16)f.y;
    u.h[2] = (__bf16)f.z; u.h[3] = (__bf16)f.w;
    out[i] = u.u;
  }
}

// NT GEMM: y[m][n] = sum_k A[m][k] * W[n][k] + bias[n], scaled. W is bf16.
// MODE 0: A f32, store bf16 [B,H,S,DH]   (Q, K)
// MODE 1: A f32, store bf16 [B,H,DH,S]   (V transposed)
// MODE 2: A bf16, store f32 [M, 768]     (output projection)
// M = 8192, N = 768, K = 768. Block = 8 waves; wave tile 16(M) x 64(N).
template <int MODE>
__global__ __launch_bounds__(256) void gemm_nt(const float* __restrict__ Af,
                                               const __bf16* __restrict__ Ab,
                                               const __bf16* __restrict__ W,
                                               const float* __restrict__ bias,
                                               float scale,
                                               float* __restrict__ outf,
                                               __bf16* __restrict__ outb) {
  const int lane = threadIdx.x & 31;
  const int wave = threadIdx.x >> 5;
  const int l16  = lane & 15;
  const int hl   = lane >> 4;           // lane half: 0 or 1
  const int m0   = blockIdx.x * 128 + wave * 16;
  const int n0   = blockIdx.y * 64;

  v8f c[4] = {};
  const float*  arowf = Af + (size_t)(m0 + l16) * DD;
  const __bf16* arowb = Ab + (size_t)(m0 + l16) * DD;

  for (int kk = 0; kk < DD; kk += 32) {
    BF16x16 a;
    if (MODE == 2) {
      const __bf16* p = arowb + kk + hl * 8;
      a.q[0] = *reinterpret_cast<const uint4*>(p);
      a.q[1] = *reinterpret_cast<const uint4*>(p + 16);
    } else {
      a.v = load_cvt_A(arowf + kk + hl * 8);
    }
    #pragma unroll
    for (int nf = 0; nf < 4; ++nf) {
      const __bf16* wp = W + (size_t)(n0 + nf * 16 + l16) * DD + kk + hl * 8;
      BF16x16 b;
      b.q[0] = *reinterpret_cast<const uint4*>(wp);
      b.q[1] = *reinterpret_cast<const uint4*>(wp + 16);
      c[nf] = wmma_bf16(a.v, b.v, c[nf]);
    }
  }

  #pragma unroll
  for (int nf = 0; nf < 4; ++nf) {
    const int n  = n0 + nf * 16 + l16;
    const float bv = bias[n];
    #pragma unroll
    for (int r = 0; r < 8; ++r) {
      const int m = m0 + r + hl * 8;
      const float val = (c[nf][r] + bv) * scale;
      if (MODE == 2) {
        outf[(size_t)m * DD + n] = val;
      } else {
        const int b = m >> 11, s = m & 2047;
        const int h = n >> 6,  d = n & 63;
        if (MODE == 0)
          outb[(((size_t)(b * HH + h) * SS + s) << 6) + d] = (__bf16)val;
        else
          outb[((size_t)(b * HH + h) * DHH + d) * SS + s] = (__bf16)val;
      }
    }
  }
}

__global__ void zero_scalar(float* p) { if (threadIdx.x == 0) *p = 0.0f; }

// Flash attention, shift-softmax (no running max; fixed shift is exact for
// softmax/entropy by shift-invariance, safe for |s - SHIFT| < 87 in fp32).
// Row sums computed by WMMA against an all-ones B matrix: the result
// replicates each row sum across all columns == the C-fragment stats layout.
__global__ __launch_bounds__(256) void flash_attn(const __bf16* __restrict__ Qb,
                                                  const __bf16* __restrict__ Kb,
                                                  const __bf16* __restrict__ Vt,
                                                  __bf16* __restrict__ Obf,
                                                  float* __restrict__ loss) {
  __shared__ __bf16 pbuf[8][16][32];
  __shared__ __bf16 xbuf[8][16][32];
  const int lane = threadIdx.x & 31;
  const int wave = threadIdx.x >> 5;
  const int l16  = lane & 15;
  const int hl   = lane >> 4;
  const int b = blockIdx.z, h = blockIdx.y;
  const int q0 = (blockIdx.x * 8 + wave) * 16;
  const size_t bh = (size_t)(b * HH + h);
  const float SHIFT = 10.0f;

  v16bf onesv;
  #pragma unroll
  for (int i = 0; i < 16; ++i) onesv[i] = (__bf16)1.0f;

  // Q A-fragments for the two K(=d)-steps, loaded once (already bf16, pre-scaled).
  BF16x16 qa[2];
  {
    const __bf16* qrow = Qb + (bh * SS + q0 + l16) * DHH;
    #pragma unroll
    for (int i = 0; i < 2; ++i) {
      const __bf16* p = qrow + i * 32 + hl * 8;
      qa[i].q[0] = *reinterpret_cast<const uint4*>(p);
      qa[i].q[1] = *reinterpret_cast<const uint4*>(p + 16);
    }
  }

  v8f o[4] = {};
  v8f lsum = {};   // per-row sum of exp(s - SHIFT), replicated across columns
  v8f tsum = {};   // per-row sum of p * (s - SHIFT)

  const __bf16* Kbase = Kb + bh * (size_t)SS * DHH;
  const __bf16* Vbase = Vt + bh * (size_t)DHH * SS;

  for (int j0 = 0; j0 < SS; j0 += 32) {
    // ---- scores: 16 x 32 as two 16x16 C-fragments ----
    v8f s0 = {}, s1 = {};
    #pragma unroll
    for (int i = 0; i < 2; ++i) {   // d-chunks of 32
      BF16x16 bk;
      const __bf16* kp0 = Kbase + (size_t)(j0 + l16) * DHH + i * 32 + hl * 8;
      bk.q[0] = *reinterpret_cast<const uint4*>(kp0);
      bk.q[1] = *reinterpret_cast<const uint4*>(kp0 + 16);
      s0 = wmma_bf16(qa[i].v, bk.v, s0);
      const __bf16* kp1 = Kbase + (size_t)(j0 + 16 + l16) * DHH + i * 32 + hl * 8;
      bk.q[0] = *reinterpret_cast<const uint4*>(kp1);
      bk.q[1] = *reinterpret_cast<const uint4*>(kp1 + 16);
      s1 = wmma_bf16(qa[i].v, bk.v, s1);
    }

    // ---- p = exp(s - SHIFT), px = p*(s - SHIFT); stage both to LDS ----
    #pragma unroll
    for (int r = 0; r < 8; ++r) {
      const float x0 = s0[r] - SHIFT, x1 = s1[r] - SHIFT;
      const float p0 = __expf(x0),    p1 = __expf(x1);
      pbuf[wave][r + hl * 8][l16]      = (__bf16)p0;
      pbuf[wave][r + hl * 8][16 + l16] = (__bf16)p1;
      xbuf[wave][r + hl * 8][l16]      = (__bf16)(p0 * x0);
      xbuf[wave][r + hl * 8][16 + l16] = (__bf16)(p1 * x1);
    }

    // C-layout -> A-layout transpose through LDS (wave-private; LDS is
    // in-order within a wave, asm blocks compiler reordering)
    asm volatile("s_wait_dscnt 0x0" ::: "memory");
    __builtin_amdgcn_wave_barrier();
    BF16x16 pa, xa;
    {
      const __bf16* pp = &pbuf[wave][l16][hl * 8];
      pa.q[0] = *reinterpret_cast<const uint4*>(pp);
      pa.q[1] = *reinterpret_cast<const uint4*>(pp + 16);
      const __bf16* xp = &xbuf[wave][l16][hl * 8];
      xa.q[0] = *reinterpret_cast<const uint4*>(xp);
      xa.q[1] = *reinterpret_cast<const uint4*>(xp + 16);
    }

    // ---- row sums via ones-WMMA; O += P @ V ----
    lsum = wmma_bf16(pa.v, onesv, lsum);
    tsum = wmma_bf16(xa.v, onesv, tsum);
    #pragma unroll
    for (int f = 0; f < 4; ++f) {
      BF16x16 bv;
      const __bf16* vp = Vbase + (size_t)(f * 16 + l16) * SS + j0 + hl * 8;
      bv.q[0] = *reinterpret_cast<const uint4*>(vp);
      bv.q[1] = *reinterpret_cast<const uint4*>(vp + 16);
      o[f] = wmma_bf16(pa.v, bv.v, o[f]);
    }
  }

  // ---- epilogue: normalize, entropy, store bf16 ----
  float hsum = 0.0f;
  #pragma unroll
  for (int r = 0; r < 8; ++r) {
    const float l   = lsum[r];
    const float inv = 1.0f / l;
    #pragma unroll
    for (int f = 0; f < 4; ++f) o[f][r] *= inv;
    if (l16 == 0) hsum += logf(l) - tsum[r] * inv;
  }

  #pragma unroll
  for (int f = 0; f < 4; ++f) {
    #pragma unroll
    for (int r = 0; r < 8; ++r) {
      const int q = q0 + r + hl * 8;
      Obf[((size_t)(b * SS + q)) * DD + h * DHH + f * 16 + l16] = (__bf16)o[f][r];
    }
  }

  hsum += __shfl_xor(hsum, 16, 32);
  if (lane == 0)
    atomicAdd(loss, hsum * (1.0f / ((float)BB * HH * (float)SS * (float)SS)));
}

extern "C" void kernel_launch(void* const* d_in, const int* in_sizes, int n_in,
                              void* d_out, int out_size, void* d_ws, size_t ws_size,
                              hipStream_t stream) {
  (void)in_sizes; (void)n_in; (void)out_size; (void)ws_size;
  const float* xq = (const float*)d_in[0];
  const float* xk = (const float*)d_in[1];
  const float* xv = (const float*)d_in[2];
  const float* wq = (const float*)d_in[3];
  const float* bq = (const float*)d_in[4];
  const float* wk = (const float*)d_in[5];
  const float* bk = (const float*)d_in[6];
  const float* wv = (const float*)d_in[7];
  const float* bv = (const float*)d_in[8];
  const float* wo = (const float*)d_in[9];
  const float* bo = (const float*)d_in[10];

  float* out  = (float*)d_out;
  float* loss = out + (size_t)BB * SS * DD;

  const size_t WEL = (size_t)DD * DD;        // 589,824
  const size_t NEL = (size_t)BB * SS * DD;   // 6,291,456
  __bf16* Wqb = (__bf16*)d_ws;
  __bf16* Wkb = Wqb + WEL;
  __bf16* Wvb = Wkb + WEL;
  __bf16* Wob = Wvb + WEL;
  __bf16* Qb  = Wob + WEL;
  __bf16* Kb  = Qb + NEL;
  __bf16* Vt  = Kb + NEL;
  __bf16* Ob  = Vt + NEL;                    // bf16 attention output [B,S,D]

  // Pre-convert weights to bf16 (one conversion instead of 64x inside GEMM).
  const int n4 = (int)(WEL / 4);             // 147,456
  const int cblk = (n4 + 255) / 256;
  cvt4<<<cblk, 256, 0, stream>>>((const float4*)wq, (uint2*)Wqb, n4);
  cvt4<<<cblk, 256, 0, stream>>>((const float4*)wk, (uint2*)Wkb, n4);
  cvt4<<<cblk, 256, 0, stream>>>((const float4*)wv, (uint2*)Wvb, n4);
  cvt4<<<cblk, 256, 0, stream>>>((const float4*)wo, (uint2*)Wob, n4);

  const dim3 gblk(64, 12);                   // M/128 x N/64
  const float qscale = 0.125f;               // 1/sqrt(DH), folded into Q projection

  gemm_nt<0><<<gblk, 256, 0, stream>>>(xq, nullptr, Wqb, bq, qscale, nullptr, Qb);
  gemm_nt<0><<<gblk, 256, 0, stream>>>(xk, nullptr, Wkb, bk, 1.0f,   nullptr, Kb);
  gemm_nt<1><<<gblk, 256, 0, stream>>>(xv, nullptr, Wvb, bv, 1.0f,   nullptr, Vt);
  zero_scalar<<<1, 32, 0, stream>>>(loss);
  flash_attn<<<dim3(SS / 128, HH, BB), 256, 0, stream>>>(Qb, Kb, Vt, Ob, loss);
  gemm_nt<2><<<gblk, 256, 0, stream>>>(nullptr, Ob, Wob, bo, 1.0f, out, nullptr);
}
